// RelationalGATLinkPredictor_70076686401551
// MI455X (gfx1250) — matrix operand order repaired and checked by reading, not compiled, not run on hardware
//
#include <hip/hip_runtime.h>
#include <hip/hip_bf16.h>
#include <math.h>

#define N_ENT 50000
#define N_RELS 4
#define EMB 128
#define HID 128
#define HEADS 4
#define OUT_F 256
#define F1 (HEADS * HID)   // 512
#define NEG_SLOPE 0.2f
#define LN_EPS 1e-5f

typedef __bf16 bf16;
typedef __attribute__((ext_vector_type(16))) __bf16 v16bf;
typedef __attribute__((ext_vector_type(8)))  __bf16 v8bf;   // 16 bytes, 16B aligned
typedef __attribute__((ext_vector_type(8)))  float  v8f;

// ---------------------------------------------------------------------------
// Utility device functions
// ---------------------------------------------------------------------------
__device__ __forceinline__ float waveReduceSum(float v) {
    #pragma unroll
    for (int off = 16; off > 0; off >>= 1) v += __shfl_xor(v, off, 32);
    return v;
}

// order-preserving float atomic max (handles negatives, init = -inf)
__device__ __forceinline__ void atomicMaxF(float* addr, float v) {
    int iv = __float_as_int(v);
    if (iv >= 0) atomicMax((int*)addr, iv);
    else         atomicMin((unsigned int*)addr, (unsigned int)iv);
}

// ---------------------------------------------------------------------------
// Elementwise helpers
// ---------------------------------------------------------------------------
__global__ void cvt_f32_to_bf16(const float* __restrict__ s, bf16* __restrict__ d, long n) {
    long i = (long)blockIdx.x * 256 + threadIdx.x;
    if (i < n) d[i] = (bf16)s[i];
}

__global__ void fill_f32(float* __restrict__ p, float v, long n) {
    long i = (long)blockIdx.x * 256 + threadIdx.x;
    if (i < n) p[i] = v;
}

// Transposing f32 -> bf16 conversion for weights:  dst[N x K] = (bf16)src[K x N]^T
// 32x32 tiles via LDS; 256 threads. K, N multiples of 32 here.
__global__ void cvt_transpose_bf16(const float* __restrict__ src, bf16* __restrict__ dst,
                                   int K, int N) {
    __shared__ float t[32][33];
    int k0 = blockIdx.x * 32, n0 = blockIdx.y * 32;
    int tx = threadIdx.x & 31, ty = threadIdx.x >> 5;   // 8 rows per pass
    #pragma unroll
    for (int i = ty; i < 32; i += 8)
        t[i][tx] = src[(long)(k0 + i) * N + n0 + tx];
    __syncthreads();
    #pragma unroll
    for (int i = ty; i < 32; i += 8)
        dst[(long)(n0 + i) * K + k0 + tx] = (bf16)t[tx][i];
}

// ---------------------------------------------------------------------------
// WMMA bf16 GEMM:  C[M,N] = A[M,K] @ B[K,N] (+ bias) (+= C if accumulate)
// A bf16 row-major [M,K]; Bt bf16 row-major TRANSPOSED weights [N,K]; C f32.
// N % 64 == 0, K % 32 == 0, M arbitrary.
// Block: 128 threads (4 waves). Block tile 64x64, wave tile 16x64.
// All LDS fragment traffic is ds_load_b128 / ds_store_b128.
// ---------------------------------------------------------------------------
#define LDA 40   // bf16 elements per LDS row (80 bytes: 16B-aligned, bank-spread)
__global__ void gemm_bf16_wmma(const bf16* __restrict__ A, const bf16* __restrict__ Bt,
                               float* __restrict__ C, const float* __restrict__ bias,
                               int M, int N, int K, int accumulate) {
    __shared__ __attribute__((aligned(16))) bf16 lA[64 * LDA];  // A tile, row-major 64x32
    __shared__ __attribute__((aligned(16))) bf16 lB[64 * LDA];  // Bt tile, row-major 64x32

    const int tid  = threadIdx.x;
    const int lane = tid & 31;
    const int wv   = tid >> 5;          // 0..3
    const int bm   = blockIdx.y * 64;
    const int bn   = blockIdx.x * 64;
    const int g    = lane >> 4;         // lane half
    const int mr   = lane & 15;

    v8f acc[4];
    #pragma unroll
    for (int t = 0; t < 4; ++t)
        #pragma unroll
        for (int j = 0; j < 8; ++j) acc[t][j] = 0.0f;

    const int srow = tid >> 1;            // staging row 0..63
    const int scol = (tid & 1) * 16;      // staging col 0 or 16

    for (int kt = 0; kt < K; kt += 32) {
        // ---- stage A tile: 64 rows x 32 cols, 16B vector path ----
        {
            int grow = bm + srow;
            v8bf q0{}, q1{};
            if (grow < M) {
                const v8bf* gsrc = (const v8bf*)(A + (long)grow * K + kt + scol);
                q0 = gsrc[0];
                q1 = gsrc[1];
                if (kt + 32 < K) __builtin_prefetch((const void*)(gsrc + 4), 0, 3);
            }
            *(v8bf*)&lA[srow * LDA + scol]     = q0;
            *(v8bf*)&lA[srow * LDA + scol + 8] = q1;
        }
        // ---- stage Bt tile: 64 rows x 32 cols (rows = N cols, cols = K) ----
        {
            const v8bf* gsrc = (const v8bf*)(Bt + (long)(bn + srow) * K + kt + scol);
            v8bf q0 = gsrc[0];
            v8bf q1 = gsrc[1];
            if (kt + 32 < K) __builtin_prefetch((const void*)(gsrc + 4), 0, 3);
            *(v8bf*)&lB[srow * LDA + scol]     = q0;
            *(v8bf*)&lB[srow * LDA + scol + 8] = q1;
        }
        __syncthreads();

        // A fragment (ISA 7.12.2): lane(m,g) = A[m][g*8..+7] ++ A[m][g*8+16..+23]
        const int mbase = (wv * 16 + mr) * LDA + g * 8;
        v8bf a0 = *(const v8bf*)&lA[mbase];
        v8bf a1 = *(const v8bf*)&lA[mbase + 16];
        v16bf af = __builtin_shufflevector(a0, a1, 0, 1, 2, 3, 4, 5, 6, 7,
                                                   8, 9, 10, 11, 12, 13, 14, 15);
        #pragma unroll
        for (int nt = 0; nt < 4; ++nt) {
            // B fragment: lane(n,g) = Bt[n][g*8..+7] ++ Bt[n][g*8+16..+23]
            const int nbase = (nt * 16 + mr) * LDA + g * 8;
            v8bf b0 = *(const v8bf*)&lB[nbase];
            v8bf b1 = *(const v8bf*)&lB[nbase + 16];
            v16bf bfv = __builtin_shufflevector(b0, b1, 0, 1, 2, 3, 4, 5, 6, 7,
                                                        8, 9, 10, 11, 12, 13, 14, 15);
            acc[nt] = __builtin_amdgcn_wmma_f32_16x16x32_bf16(
                false, af, false, bfv, (short)0, acc[nt], false, false);
        }
        __syncthreads();
    }

    // store: lane%16 = N col, VGPR j = M row (j + 8*laneHalf)
    #pragma unroll
    for (int nt = 0; nt < 4; ++nt) {
        #pragma unroll
        for (int j = 0; j < 8; ++j) {
            int row = bm + wv * 16 + j + g * 8;
            int col = bn + nt * 16 + mr;
            if (row < M) {
                float v = acc[nt][j];
                if (bias) v += bias[col];
                long idx = (long)row * N + col;
                if (accumulate) v += C[idx];
                C[idx] = v;
            }
        }
    }
}

// ---------------------------------------------------------------------------
// GAT pieces
// ---------------------------------------------------------------------------
// alpha_s[n,h] = <xp[n,h,:], a_src[h,:]>, alpha_d similarly. One wave per (n,h).
__global__ void att_coef(const float* __restrict__ xp,
                         const float* __restrict__ a_src, const float* __restrict__ a_dst,
                         float* __restrict__ alpha_s, float* __restrict__ alpha_d,
                         int H, int C) {
    long gw  = ((long)blockIdx.x * blockDim.x + threadIdx.x) >> 5;
    int lane = threadIdx.x & 31;
    if (gw >= (long)N_ENT * H) return;
    int n = (int)(gw / H), h = (int)(gw % H);
    const float* row = xp + (long)n * H * C + (long)h * C;
    float s = 0.f, d = 0.f;
    for (int c = lane; c < C; c += 32) {
        float x = row[c];
        s += x * a_src[h * C + c];
        d += x * a_dst[h * C + c];
    }
    s = waveReduceSum(s);
    d = waveReduceSum(d);
    if (lane == 0) { alpha_s[n * H + h] = s; alpha_d[n * H + h] = d; }
}

// edge logits + segment max.  Edges e < nE come from arrays; e >= nE are self loops.
__global__ void edge_logit_max(const int* __restrict__ src, const int* __restrict__ dst,
                               int nE, long nTotH, int H,
                               const float* __restrict__ alpha_s, const float* __restrict__ alpha_d,
                               float* __restrict__ ebuf, float* __restrict__ nmax) {
    long idx = (long)blockIdx.x * 256 + threadIdx.x;
    if (idx >= nTotH) return;
    int e = (int)(idx / H), h = (int)(idx % H);
    int s, dn;
    if (e < nE) { s = src[e]; dn = dst[e]; } else { s = dn = e - nE; }
    float v = alpha_s[s * H + h] + alpha_d[dn * H + h];
    v = v > 0.f ? v : NEG_SLOPE * v;
    ebuf[idx] = v;
    atomicMaxF(nmax + dn * H + h, v);
}

// exp(e - max) + segment sum
__global__ void edge_exp_sum(const int* __restrict__ dst, int nE, long nTotH, int H,
                             const float* __restrict__ nmax,
                             float* __restrict__ ebuf, float* __restrict__ den) {
    long idx = (long)blockIdx.x * 256 + threadIdx.x;
    if (idx >= nTotH) return;
    int e = (int)(idx / H), h = (int)(idx % H);
    int dn = (e < nE) ? dst[e] : e - nE;
    float ee = __expf(ebuf[idx] - nmax[dn * H + h]);
    ebuf[idx] = ee;
    atomicAdd(den + dn * H + h, ee);
}

// out[dst,h,:] += xp[src,h,:] * alpha.  One wave per (edge, head).
__global__ void edge_aggregate(const int* __restrict__ src, const int* __restrict__ dst,
                               int nE, long nTotH, int H, int C,
                               const float* __restrict__ xp, const float* __restrict__ ebuf,
                               const float* __restrict__ den, float* __restrict__ outp) {
    long gw  = ((long)blockIdx.x * 256 + threadIdx.x) >> 5;
    int lane = threadIdx.x & 31;
    if (gw >= nTotH) return;
    int e = (int)(gw / H), h = (int)(gw % H);
    int s, dn;
    if (e < nE) { s = src[e]; dn = dst[e]; } else { s = dn = e - nE; }
    float a = ebuf[gw] / (den[dn * H + h] + 1e-16f);
    const float* xr   = xp + (long)s * H * C + (long)h * C;
    float*       orow = outp + (long)dn * H * C + (long)h * C;
    for (int c = lane; c < C; c += 32) atomicAdd(orow + c, xr[c] * a);
}

// x_acc += elu(agg + bias)
__global__ void elu_acc(const float* __restrict__ agg, const float* __restrict__ bias,
                        float* __restrict__ xacc, int F, long n) {
    long i = (long)blockIdx.x * 256 + threadIdx.x;
    if (i >= n) return;
    float v = agg[i] + bias[(int)(i % F)];
    v = v > 0.f ? v : (__expf(v) - 1.f);
    xacc[i] += v;
}

__global__ void bias_sum(const float* __restrict__ b2s, float* __restrict__ b2sum) {
    int c = threadIdx.x;  // 256 threads
    float s = 0.f;
    #pragma unroll
    for (int r = 0; r < N_RELS; ++r) s += b2s[r * OUT_F + c];
    b2sum[c] = s;
}

// LayerNorm over OUT_F=256, one wave per node (8 elems per lane)
__global__ void layer_norm(const float* __restrict__ y, const float* __restrict__ b2sum,
                           const float* __restrict__ lng, const float* __restrict__ lnb,
                           float* __restrict__ ent) {
    long gw  = ((long)blockIdx.x * blockDim.x + threadIdx.x) >> 5;
    int lane = threadIdx.x & 31;
    if (gw >= N_ENT) return;
    const float* row = y + gw * OUT_F;
    float vals[8];
    float s = 0.f;
    #pragma unroll
    for (int j = 0; j < 8; ++j) {
        int c = lane + 32 * j;
        vals[j] = row[c] + b2sum[c];
        s += vals[j];
    }
    s = waveReduceSum(s);
    float mu = s * (1.f / OUT_F);
    float vs = 0.f;
    #pragma unroll
    for (int j = 0; j < 8; ++j) { float d = vals[j] - mu; vs += d * d; }
    vs = waveReduceSum(vs);
    float rstd = rsqrtf(vs * (1.f / OUT_F) + LN_EPS);
    float* orow = ent + gw * OUT_F;
    #pragma unroll
    for (int j = 0; j < 8; ++j) {
        int c = lane + 32 * j;
        orow[c] = (vals[j] - mu) * rstd * lng[c] + lnb[c];
    }
}

// DistMult scoring, one wave per triple
__global__ void distmult(const int* __restrict__ th, const int* __restrict__ tr,
                         const int* __restrict__ tt, const float* __restrict__ ent,
                         const float* __restrict__ rel_emb, float* __restrict__ out, int B) {
    long gw  = ((long)blockIdx.x * blockDim.x + threadIdx.x) >> 5;
    int lane = threadIdx.x & 31;
    if (gw >= B) return;
    const float* eh = ent + (long)th[gw] * OUT_F;
    const float* et = ent + (long)tt[gw] * OUT_F;
    const float* re = rel_emb + (long)tr[gw] * OUT_F;
    float s = 0.f;
    for (int c = lane; c < OUT_F; c += 32) s += eh[c] * re[c] * et[c];
    s = waveReduceSum(s);
    if (lane == 0) out[gw] = s;
}

// ---------------------------------------------------------------------------
// Host launch
// ---------------------------------------------------------------------------
static inline long cdivl(long a, long b) { return (a + b - 1) / b; }

extern "C" void kernel_launch(void* const* d_in, const int* in_sizes, int n_in,
                              void* d_out, int out_size, void* d_ws, size_t ws_size,
                              hipStream_t stream) {
    const int*   edges      = (const int*)d_in[0];
    const int*   trip_h     = (const int*)d_in[1];
    const int*   trip_r     = (const int*)d_in[2];
    const int*   trip_t     = (const int*)d_in[3];
    const float* entity_emb = (const float*)d_in[4];
    const float* W1s        = (const float*)d_in[5];
    const float* as1        = (const float*)d_in[6];
    const float* ad1        = (const float*)d_in[7];
    const float* b1s        = (const float*)d_in[8];
    const float* W2s        = (const float*)d_in[9];
    const float* as2        = (const float*)d_in[10];
    const float* ad2        = (const float*)d_in[11];
    const float* b2s        = (const float*)d_in[12];
    const float* resW       = (const float*)d_in[13];
    const float* resb       = (const float*)d_in[14];
    const float* lng        = (const float*)d_in[15];
    const float* lnb        = (const float*)d_in[16];
    const float* rel_emb    = (const float*)d_in[17];

    const int  E    = in_sizes[0] / (N_RELS * 2);
    const int  B    = in_sizes[1];
    const long nTot = (long)E + N_ENT;

    // bump allocator over d_ws
    char* p = (char*)d_ws;
    auto alloc = [&](size_t bytes) -> void* {
        void* r = (void*)p;
        p += (bytes + 255) & ~(size_t)255;
        return r;
    };
    bf16*  x0b     = (bf16*)alloc((size_t)N_ENT * EMB * sizeof(bf16));
    bf16*  W1b     = (bf16*)alloc((size_t)N_RELS * EMB * F1 * sizeof(bf16));    // [F1 x EMB] per rel
    bf16*  W2b     = (bf16*)alloc((size_t)N_RELS * F1 * OUT_F * sizeof(bf16));  // [OUT_F x F1] per rel
    bf16*  resWb   = (bf16*)alloc((size_t)EMB * OUT_F * sizeof(bf16));          // [OUT_F x EMB]
    float* xp      = (float*)alloc((size_t)N_ENT * F1 * sizeof(float));
    float* agg     = (float*)alloc((size_t)N_ENT * F1 * sizeof(float));
    float* x_acc   = (float*)alloc((size_t)N_ENT * F1 * sizeof(float));
    bf16*  x_b     = (bf16*)alloc((size_t)N_ENT * F1 * sizeof(bf16));
    float* y_acc   = (float*)alloc((size_t)N_ENT * OUT_F * sizeof(float));
    float* entv    = (float*)alloc((size_t)N_ENT * OUT_F * sizeof(float));
    float* alpha_s = (float*)alloc((size_t)N_ENT * HEADS * sizeof(float));
    float* alpha_d = (float*)alloc((size_t)N_ENT * HEADS * sizeof(float));
    float* nmax    = (float*)alloc((size_t)N_ENT * HEADS * sizeof(float));
    float* den     = (float*)alloc((size_t)N_ENT * HEADS * sizeof(float));
    float* ebuf    = (float*)alloc((size_t)nTot * HEADS * sizeof(float));
    float* b2sum   = (float*)alloc((size_t)OUT_F * sizeof(float));

    #define LAUNCH_EW(kern, n, ...) \
        kern<<<dim3((unsigned)cdivl((long)(n), 256)), dim3(256), 0, stream>>>(__VA_ARGS__)

    // --- precision conversion: activations plain, weights transposed ---
    LAUNCH_EW(cvt_f32_to_bf16, (long)N_ENT * EMB, entity_emb, x0b, (long)N_ENT * EMB);
    for (int r = 0; r < N_RELS; ++r) {
        cvt_transpose_bf16<<<dim3(EMB / 32, F1 / 32), dim3(256), 0, stream>>>(
            W1s + (long)r * EMB * F1, W1b + (long)r * F1 * EMB, EMB, F1);
        cvt_transpose_bf16<<<dim3(F1 / 32, OUT_F / 32), dim3(256), 0, stream>>>(
            W2s + (long)r * F1 * OUT_F, W2b + (long)r * OUT_F * F1, F1, OUT_F);
    }
    cvt_transpose_bf16<<<dim3(EMB / 32, OUT_F / 32), dim3(256), 0, stream>>>(
        resW, resWb, EMB, OUT_F);

    LAUNCH_EW(fill_f32, (long)N_ENT * F1, x_acc, 0.f, (long)N_ENT * F1);

    const dim3 gemmBlk(128);
    // ================= Layer 1 (per relation) =================
    for (int r = 0; r < N_RELS; ++r) {
        const int* src = edges + ((long)r * 2 + 0) * E;
        const int* dst = edges + ((long)r * 2 + 1) * E;
        const long nTotH = nTot * HEADS;

        gemm_bf16_wmma<<<dim3(F1 / 64, (unsigned)cdivl(N_ENT, 64)), gemmBlk, 0, stream>>>(
            x0b, W1b + (long)r * F1 * EMB, xp, nullptr, N_ENT, F1, EMB, 0);

        att_coef<<<dim3((unsigned)cdivl((long)N_ENT * HEADS * 32, 256)), dim3(256), 0, stream>>>(
            xp, as1 + (long)r * HEADS * HID, ad1 + (long)r * HEADS * HID,
            alpha_s, alpha_d, HEADS, HID);

        LAUNCH_EW(fill_f32, (long)N_ENT * HEADS, nmax, -INFINITY, (long)N_ENT * HEADS);
        LAUNCH_EW(fill_f32, (long)N_ENT * HEADS, den, 0.f, (long)N_ENT * HEADS);

        LAUNCH_EW(edge_logit_max, nTotH, src, dst, E, nTotH, HEADS, alpha_s, alpha_d, ebuf, nmax);
        LAUNCH_EW(edge_exp_sum, nTotH, dst, E, nTotH, HEADS, nmax, ebuf, den);

        LAUNCH_EW(fill_f32, (long)N_ENT * F1, agg, 0.f, (long)N_ENT * F1);
        edge_aggregate<<<dim3((unsigned)cdivl(nTotH * 32, 256)), dim3(256), 0, stream>>>(
            src, dst, E, nTotH, HEADS, HID, xp, ebuf, den, agg);

        LAUNCH_EW(elu_acc, (long)N_ENT * F1, agg, b1s + (long)r * F1, x_acc, F1, (long)N_ENT * F1);
    }

    // ================= Layer 2 (per relation) =================
    LAUNCH_EW(cvt_f32_to_bf16, (long)N_ENT * F1, x_acc, x_b, (long)N_ENT * F1);
    LAUNCH_EW(fill_f32, (long)N_ENT * OUT_F, y_acc, 0.f, (long)N_ENT * OUT_F);

    for (int r = 0; r < N_RELS; ++r) {
        const int* src = edges + ((long)r * 2 + 0) * E;
        const int* dst = edges + ((long)r * 2 + 1) * E;
        const long nTotH = nTot;  // H = 1

        gemm_bf16_wmma<<<dim3(OUT_F / 64, (unsigned)cdivl(N_ENT, 64)), gemmBlk, 0, stream>>>(
            x_b, W2b + (long)r * OUT_F * F1, xp, nullptr, N_ENT, OUT_F, F1, 0);

        att_coef<<<dim3((unsigned)cdivl((long)N_ENT * 32, 256)), dim3(256), 0, stream>>>(
            xp, as2 + (long)r * OUT_F, ad2 + (long)r * OUT_F, alpha_s, alpha_d, 1, OUT_F);

        LAUNCH_EW(fill_f32, (long)N_ENT, nmax, -INFINITY, (long)N_ENT);
        LAUNCH_EW(fill_f32, (long)N_ENT, den, 0.f, (long)N_ENT);

        LAUNCH_EW(edge_logit_max, nTotH, src, dst, E, nTotH, 1, alpha_s, alpha_d, ebuf, nmax);
        LAUNCH_EW(edge_exp_sum, nTotH, dst, E, nTotH, 1, nmax, ebuf, den);

        // accumulate all relations directly into y_acc (no nonlinearity between)
        edge_aggregate<<<dim3((unsigned)cdivl(nTotH * 32, 256)), dim3(256), 0, stream>>>(
            src, dst, E, nTotH, 1, OUT_F, xp, ebuf, den, y_acc);
    }

    // residual: y_acc += x0 @ resW + resb
    gemm_bf16_wmma<<<dim3(OUT_F / 64, (unsigned)cdivl(N_ENT, 64)), gemmBlk, 0, stream>>>(
        x0b, resWb, y_acc, resb, N_ENT, OUT_F, EMB, 1);

    // sum of layer-2 biases across relations, applied inside LayerNorm
    bias_sum<<<dim3(1), dim3(OUT_F), 0, stream>>>(b2s, b2sum);

    layer_norm<<<dim3((unsigned)cdivl((long)N_ENT * 32, 256)), dim3(256), 0, stream>>>(
        y_acc, b2sum, lng, lnb, entv);

    distmult<<<dim3((unsigned)cdivl((long)B * 32, 256)), dim3(256), 0, stream>>>(
        trip_h, trip_r, trip_t, entv, rel_emb, (float*)d_out, B);

    #undef LAUNCH_EW
}
